// LOBRM_84155589198295
// MI455X (gfx1250) — matrix operand order repaired and checked by reading, not compiled, not run on hardware
//
#include <hip/hip_runtime.h>
#include <hip/hip_bf16.h>
#include <math.h>

#define BB   4096
#define TT   64
#define DD   100
#define FF   50
#define LL   64
#define KK   128
#define GIC  114      // L + F = 64 + 50
#define TB   64       // batch rows per block

#define LDA  136      // f16 stride for 128-wide buffers (+8 pad, conflict-free)
#define LDH  72       // f16 stride for 64-wide buffers (+8 pad)
#define LDF  68       // f32 stride for 64-wide buffers (+4 pad)

typedef __attribute__((ext_vector_type(16))) _Float16 v16h;
typedef __attribute__((ext_vector_type(8)))  float    v8f;

// Branch-free transcendentals (one v_exp_f32 each; saturate correctly via inf/0)
__device__ __forceinline__ float fast_tanh(float x) {
  const float e = __expf(2.f * x);       // +inf for large x, 0 for very negative
  return 1.f - 2.f / (e + 1.f);          // -> +1 / -1 at the extremes
}
__device__ __forceinline__ float fast_sigmoid(float x) {
  return 1.f / (1.f + __expf(-x));
}

// ---------------------------------------------------------------------------
// 16x64 band GEMM: acc[4] (four 16x16 f32 tiles) += A(16xKDIM f16) * W(KDIMx64)
// A row-major f16 LDS stride SA; W stored TRANSPOSED [n][k] stride SW so both
// fragments are contiguous ds_load_b128 pairs.
// ---------------------------------------------------------------------------
template<int KDIM, int SA, int SW>
__device__ __forceinline__ void mmband(v8f acc[4], const _Float16* A,
                                       const _Float16* Wt, int row0, int lane) {
  const int ln = lane & 15;
  const int lh = lane >> 4;
#pragma unroll
  for (int n = 0; n < 4; ++n) { v8f z = {0,0,0,0,0,0,0,0}; acc[n] = z; }
#pragma unroll
  for (int k0 = 0; k0 < KDIM; k0 += 32) {
    // A fragment: lane = M, 16 contiguous K halves; upper half-wave at K+16.
    v16h a;
    const _Float16* pa = A + (row0 + ln) * SA + k0 + (lh << 4);
#pragma unroll
    for (int i = 0; i < 16; ++i) a[i] = pa[i];
#pragma unroll
    for (int n = 0; n < 4; ++n) {
      // B fragment: lane = N, 16 contiguous K halves; upper half-wave at K+16.
      v16h b;
      const _Float16* pb = Wt + (n * 16 + ln) * SW + k0 + (lh << 4);
#pragma unroll
      for (int i = 0; i < 16; ++i) b[i] = pb[i];
      acc[n] = __builtin_amdgcn_wmma_f32_16x16x32_f16(
          false, a, false, b, (short)0, acc[n], false, false);
    }
  }
}

// ---------------------------------------------------------------------------
// Kernel 0: presence / exclusive-cumsum -> idx_all, upd_all (k-major) and
// time_intervals. One thread per batch row (two-pointer over sorted ts).
// ---------------------------------------------------------------------------
__global__ void k_prep(const int* __restrict__ ts, int* __restrict__ idxws,
                       float* __restrict__ updws, float* __restrict__ out_ti) {
  const int b = blockIdx.x * blockDim.x + threadIdx.x;
  if (b >= BB) return;
  const int* t = ts + b * TT;
  int p = 0, cnt = 0;
  for (int k = 0; k < KK; ++k) {
    while (p < TT && t[p] < k) ++p;
    const int present = (p < TT && t[p] == k) ? 1 : 0;
    idxws[k * BB + b] = (cnt < TT) ? cnt : TT;   // clip(counts, 0, T)
    updws[k * BB + b] = (float)present;
    cnt += present;
  }
  for (int i = 0; i < TT - 1; ++i)
    out_ti[b * (TT - 1) + i] = (float)(t[i + 1] - t[i]);
}

// ---------------------------------------------------------------------------
// Kernel 1: the K=128 step scan. 64 blocks x 128 threads (4 waves).
// ---------------------------------------------------------------------------
__global__ void __launch_bounds__(128)
k_scan(const float* __restrict__ data,
       const int* __restrict__ idxws, const float* __restrict__ updws,
       float* __restrict__ volws, float* __restrict__ out_prevy,
       const float* Wu1, const float* bu1, const float* Wu2, const float* bu2,
       const float* Wr1, const float* br1, const float* Wr2, const float* br2,
       const float* Wn1, const float* bn1, const float* Wn2, const float* bn2,
       const float* Wo1, const float* bo1, const float* Wo2, const float* bo2,
       const float* Wd,  const float* bd) {
  // --- LDS (~173 KB; WGP allows 320 KB) ---
  __shared__ _Float16 sW1t[3][64][LDA];   // Wu1^T, Wr1^T, Wn1^T  (K padded 114->128)
  __shared__ _Float16 sW2t[5][64][LDH];   // Wo1^T, Wo2^T, Wu2^T, Wr2^T, Wn2^T
  __shared__ float    sBias[8][64];       // bo1 bo2 bu1 bu2 br1 br2 bn1 bn2
  __shared__ float    sWd[64];
  __shared__ float    sBdv;
  __shared__ float    sY  [64][LDF];      // prev_y f32 (master state)
  __shared__ _Float16 sYh [64][LDH];      // prev_y f16 (GEMM input)
  __shared__ _Float16 sY1h[64][LDH];      // y1 f16
  __shared__ _Float16 sU  [64][LDH];      // u gate
  __shared__ _Float16 sH  [64][LDH];      // MLP hidden (tanh)
  __shared__ _Float16 sYC [64][LDA];      // yc / cc : [y1|feat|0-pad] f16
  __shared__ float    sUpd[64];
  __shared__ float    sM  [64];

  const int tid  = threadIdx.x;
  const int wave = tid >> 5;
  const int lane = tid & 31;
  const int ln   = lane & 15;
  const int lh   = lane >> 4;
  const int row0 = wave * 16;         // M band
  const int rb   = row0 + (lh << 3);  // C-tile row base for this half-wave

  // ---- load weights transposed -> f16 LDS ----
  for (int idx = tid; idx < 3 * 64 * 128; idx += 128) {
    const int m = idx >> 13, rem = idx & 8191, n = rem >> 7, kk = rem & 127;
    const float* W = (m == 0) ? Wu1 : (m == 1) ? Wr1 : Wn1;
    sW1t[m][n][kk] = (_Float16)((kk < GIC) ? W[kk * 64 + n] : 0.f);
  }
  for (int idx = tid; idx < 5 * 64 * 64; idx += 128) {
    const int m = idx >> 12, rem = idx & 4095, n = rem >> 6, kk = rem & 63;
    const float* W = (m == 0) ? Wo1 : (m == 1) ? Wo2 : (m == 2) ? Wu2
                   : (m == 3) ? Wr2 : Wn2;
    sW2t[m][n][kk] = (_Float16)W[kk * 64 + n];
  }
  for (int idx = tid; idx < 8 * 64; idx += 128) {
    const int j = idx >> 6, n = idx & 63;
    const float* bb = (j == 0) ? bo1 : (j == 1) ? bo2 : (j == 2) ? bu1
                    : (j == 3) ? bu2 : (j == 4) ? br1 : (j == 5) ? br2
                    : (j == 6) ? bn1 : bn2;
    sBias[j][n] = bb[n];
  }
  if (tid < 64) sWd[tid] = Wd[tid];
  if (tid == 0) sBdv = bd[0];
  for (int idx = tid; idx < 64 * LDA; idx += 128) (&sYC[0][0])[idx] = (_Float16)0.f;
  for (int idx = tid; idx < 64 * LDH; idx += 128) (&sYh[0][0])[idx] = (_Float16)0.f;
  for (int idx = tid; idx < 64 * LDF; idx += 128) (&sY[0][0])[idx]  = 0.f;
  __syncthreads();

  v8f acc[4];

  for (int step = 0; step < KK; ++step) {
    // ---- scalar phase: vol = prev_y @ Wd + bd, feature gather, mask ----
    if (tid < 64) {
      const int b     = blockIdx.x * TB + tid;
      const float upd = updws[step * BB + b];
      const int   ix  = idxws[step * BB + b];
      sUpd[tid] = upd;
      float v = sBdv;
      for (int j = 0; j < 64; ++j) v += sY[tid][j] * sWd[j];
      volws[step * BB + b] = v;
      float msum = 0.f;
      if (upd > 0.f && ix < TT) {
        const float* rowp = data + ((size_t)b * TT + ix) * DD;
        for (int j = 0; j < FF; ++j) sYC[tid][64 + j] = (_Float16)rowp[j];
        for (int j = FF; j < DD; ++j) msum += rowp[j];
      } else {
        for (int j = 0; j < FF; ++j) sYC[tid][64 + j] = (_Float16)0.f;
      }
      sM[tid] = (msum > 0.f) ? 1.f : 0.f;
    }
    __syncthreads();

    // ---- G1: h = tanh(prev_y @ Wo1 + bo1) ----
    mmband<64, LDH, LDH>(acc, &sYh[0][0], &sW2t[0][0][0], row0, lane);
#pragma unroll
    for (int n = 0; n < 4; ++n) {
      const int col = n * 16 + ln;
      const float bv = sBias[0][col];
#pragma unroll
      for (int i = 0; i < 8; ++i)
        sH[rb + i][col] = (_Float16)fast_tanh(acc[n][i] + bv);
    }
    // ---- G2: y1 = prev_y + h @ Wo2 + bo2 ----
    mmband<64, LDH, LDH>(acc, &sH[0][0], &sW2t[1][0][0], row0, lane);
#pragma unroll
    for (int n = 0; n < 4; ++n) {
      const int col = n * 16 + ln;
      const float bv = sBias[1][col];
#pragma unroll
      for (int i = 0; i < 8; ++i) {
        const float y1 = acc[n][i] + bv + sY[rb + i][col];
        sY1h[rb + i][col] = (_Float16)y1;
        sYC [rb + i][col] = (_Float16)y1;     // yc[:, :64]
      }
    }
    __syncthreads();

    // ---- u = sigmoid(tanh(yc @ Wu1 + bu1) @ Wu2 + bu2) ----
    mmband<128, LDA, LDA>(acc, &sYC[0][0], &sW1t[0][0][0], row0, lane);
#pragma unroll
    for (int n = 0; n < 4; ++n) {
      const int col = n * 16 + ln;
      const float bv = sBias[2][col];
#pragma unroll
      for (int i = 0; i < 8; ++i)
        sH[rb + i][col] = (_Float16)fast_tanh(acc[n][i] + bv);
    }
    mmband<64, LDH, LDH>(acc, &sH[0][0], &sW2t[2][0][0], row0, lane);
#pragma unroll
    for (int n = 0; n < 4; ++n) {
      const int col = n * 16 + ln;
      const float bv = sBias[3][col];
#pragma unroll
      for (int i = 0; i < 8; ++i)
        sU[rb + i][col] = (_Float16)fast_sigmoid(acc[n][i] + bv);
    }

    // ---- r = sigmoid(tanh(yc @ Wr1 + br1) @ Wr2 + br2); cc[:, :64] = y1*r ----
    mmband<128, LDA, LDA>(acc, &sYC[0][0], &sW1t[1][0][0], row0, lane);
#pragma unroll
    for (int n = 0; n < 4; ++n) {
      const int col = n * 16 + ln;
      const float bv = sBias[4][col];
#pragma unroll
      for (int i = 0; i < 8; ++i)
        sH[rb + i][col] = (_Float16)fast_tanh(acc[n][i] + bv);
    }
    mmband<64, LDH, LDH>(acc, &sH[0][0], &sW2t[3][0][0], row0, lane);
#pragma unroll
    for (int n = 0; n < 4; ++n) {
      const int col = n * 16 + ln;
      const float bv = sBias[5][col];
#pragma unroll
      for (int i = 0; i < 8; ++i) {
        const float r  = fast_sigmoid(acc[n][i] + bv);
        const float y1 = (float)sY1h[rb + i][col];
        sYC[rb + i][col] = (_Float16)(y1 * r);   // cc (feat cols unchanged)
      }
    }

    // ---- ns = tanh(cc @ Wn1 + bn1) @ Wn2 + bn2; gated state update ----
    mmband<128, LDA, LDA>(acc, &sYC[0][0], &sW1t[2][0][0], row0, lane);
#pragma unroll
    for (int n = 0; n < 4; ++n) {
      const int col = n * 16 + ln;
      const float bv = sBias[6][col];
#pragma unroll
      for (int i = 0; i < 8; ++i)
        sH[rb + i][col] = (_Float16)fast_tanh(acc[n][i] + bv);
    }
    mmband<64, LDH, LDH>(acc, &sH[0][0], &sW2t[4][0][0], row0, lane);
#pragma unroll
    for (int n = 0; n < 4; ++n) {
      const int col = n * 16 + ln;
      const float bv = sBias[7][col];
#pragma unroll
      for (int i = 0; i < 8; ++i) {
        const int row = rb + i;
        const float ns  = acc[n][i] + bv;
        const float u   = (float)sU[row][col];
        const float y1  = (float)sY1h[row][col];
        float ny        = (1.f - u) * ns + u * y1;
        const float m   = sM[row];
        ny              = m * ny + (1.f - m) * y1;
        const float upd = sUpd[row];
        const float py  = ny * upd + y1 * (1.f - upd);
        sY [row][col] = py;
        sYh[row][col] = (_Float16)py;
      }
    }
    __syncthreads();
  }

  // ---- write final prev_y ----
  for (int idx = tid; idx < 64 * 64; idx += 128) {
    const int r = idx >> 6, c = idx & 63;
    out_prevy[((size_t)blockIdx.x * TB + r) * LL + c] = sY[r][c];
  }
}

// ---------------------------------------------------------------------------
// Kernel 2: selected_state[b,t] = volume_dt[b, 1 + clip(ts[b,t],0,K-2)]
// ---------------------------------------------------------------------------
__global__ void k_select(const int* __restrict__ ts,
                         const float* __restrict__ volws,
                         float* __restrict__ out_sel) {
  const int i = blockIdx.x * blockDim.x + threadIdx.x;
  if (i >= BB * (TT - 1)) return;
  const int b = i / (TT - 1), t = i % (TT - 1);
  int inv = ts[b * TT + t];
  if (inv < 0) inv = 0;
  if (inv > KK - 2) inv = KK - 2;
  out_sel[i] = volws[(inv + 1) * BB + b];
}

extern "C" void kernel_launch(void* const* d_in, const int* in_sizes, int n_in,
                              void* d_out, int out_size, void* d_ws, size_t ws_size,
                              hipStream_t stream) {
  const float* data = (const float*)d_in[0];
  const int*   ts   = (const int*)d_in[1];
  const float* Wu1 = (const float*)d_in[2];  const float* bu1 = (const float*)d_in[3];
  const float* Wu2 = (const float*)d_in[4];  const float* bu2 = (const float*)d_in[5];
  const float* Wr1 = (const float*)d_in[6];  const float* br1 = (const float*)d_in[7];
  const float* Wr2 = (const float*)d_in[8];  const float* br2 = (const float*)d_in[9];
  const float* Wn1 = (const float*)d_in[10]; const float* bn1 = (const float*)d_in[11];
  const float* Wn2 = (const float*)d_in[12]; const float* bn2 = (const float*)d_in[13];
  const float* Wo1 = (const float*)d_in[14]; const float* bo1 = (const float*)d_in[15];
  const float* Wo2 = (const float*)d_in[16]; const float* bo2 = (const float*)d_in[17];
  const float* Wd  = (const float*)d_in[18]; const float* bd  = (const float*)d_in[19];

  int*   idxws = (int*)d_ws;
  float* updws = (float*)((char*)d_ws + (size_t)KK * BB * 4);
  float* volws = (float*)((char*)d_ws + (size_t)2 * KK * BB * 4);

  float* out_prevy = (float*)d_out;
  float* out_sel   = out_prevy + (size_t)BB * LL;
  float* out_ti    = out_sel + (size_t)BB * (TT - 1);

  hipLaunchKernelGGL(k_prep, dim3(BB / 128), dim3(128), 0, stream,
                     ts, idxws, updws, out_ti);
  hipLaunchKernelGGL(k_scan, dim3(BB / TB), dim3(128), 0, stream,
                     data, idxws, updws, volws, out_prevy,
                     Wu1, bu1, Wu2, bu2, Wr1, br1, Wr2, br2,
                     Wn1, bn1, Wn2, bn2, Wo1, bo1, Wo2, bo2, Wd, bd);
  hipLaunchKernelGGL(k_select, dim3((BB * (TT - 1) + 255) / 256), dim3(256),
                     0, stream, ts, volws, out_sel);
}
